// GATModel_71554155151375
// MI455X (gfx1250) — matrix operand order repaired and checked by reading, not compiled, not run on hardware
//
#include <hip/hip_runtime.h>
#include <hip/hip_bf16.h>

#define N_NODES 50000
#define N_EDGES 1600000
#define G_GRAPHS 64
#define NODE_IN 32
#define EDGE_IN 16
#define HID 64
#define NHEAD 4
#define NLAYERS 3
#define ROW_TILES (N_NODES / 16)   // 3125, exact

typedef __attribute__((ext_vector_type(16))) _Float16 v16h;
typedef __attribute__((ext_vector_type(8)))  float    v8f;

// ---------------------------------------------------------------------------
// float atomic max via sign-aware integer atomics (init must be -inf)
// ---------------------------------------------------------------------------
__device__ __forceinline__ void atomicMaxFloat(float* addr, float val) {
    if (val >= 0.0f) atomicMax((int*)addr, __float_as_int(val));
    else             atomicMin((unsigned int*)addr, __float_as_uint(val));
}

// ---------------------------------------------------------------------------
// h = relu(x @ Wn + bn)   x:[N,32] Wn:[32,64]
// one wave per 16-row tile; 4 col-tiles; K=32 -> 1 WMMA per col-tile
// ---------------------------------------------------------------------------
__global__ void input_gemm(const float* __restrict__ x,
                           const float* __restrict__ Wn,
                           const float* __restrict__ bn,
                           float* __restrict__ h) {
    int wave = threadIdx.x >> 5;
    int lane = threadIdx.x & 31;
    int tileM = blockIdx.x * 4 + wave;
    if (tileM >= ROW_TILES) return;            // wave-uniform: EXEC all-ones inside

    int rowLocal = lane & 15;
    int row  = tileM * 16 + rowLocal;
    int kOff = (lane < 16) ? 0 : 8;            // A layout: 16-bit 16x32 (ISA 7.12.2)
    v16h a;
    #pragma unroll
    for (int j = 0; j < 8; ++j) {
        a[j]     = (_Float16)x[row * NODE_IN + kOff + j];
        a[8 + j] = (_Float16)x[row * NODE_IN + 16 + kOff + j];
    }
    int kb = (lane < 16) ? 0 : 8;              // B layout: 32x16, same K order as A
    #pragma unroll
    for (int nt = 0; nt < 4; ++nt) {
        int col = nt * 16 + rowLocal;
        v16h b;
        #pragma unroll
        for (int j = 0; j < 16; ++j)
            b[j] = (_Float16)Wn[(kb + (j & 7) + ((j >> 3) << 4)) * HID + col];
        v8f c = {};
        c = __builtin_amdgcn_wmma_f32_16x16x32_f16(false, a, false, b,
                                                   (short)0, c, false, false);
        float bias = bn[col];
        #pragma unroll
        for (int i = 0; i < 8; ++i) {
            int m = tileM * 16 + ((lane < 16) ? i : 8 + i);
            h[m * HID + col] = fmaxf(c[i] + bias, 0.0f);
        }
    }
}

// ---------------------------------------------------------------------------
// Fused q/k/v/skip node GEMMs for one layer. Block = 4 waves; wave w computes
// matrix w for one 16-row slab (A regs shared via cache). K=64 -> 2 WMMA/tile.
// agg is initialized with the skip branch (h @ Ws + bs).
// ---------------------------------------------------------------------------
__global__ void qkvs_gemm(const float* __restrict__ h,
                          const float* __restrict__ Wq, const float* __restrict__ Wk,
                          const float* __restrict__ Wv, const float* __restrict__ Ws,
                          const float* __restrict__ bq, const float* __restrict__ bk,
                          const float* __restrict__ bv, const float* __restrict__ bs,
                          float* __restrict__ qn, float* __restrict__ kn,
                          float* __restrict__ vn, float* __restrict__ agg) {
    int wave = threadIdx.x >> 5;
    int lane = threadIdx.x & 31;
    int tileM = blockIdx.x;                    // exactly ROW_TILES blocks

    const float* W; const float* bias; float* out;
    switch (wave) {
        case 0:  W = Wq; bias = bq; out = qn;  break;
        case 1:  W = Wk; bias = bk; out = kn;  break;
        case 2:  W = Wv; bias = bv; out = vn;  break;
        default: W = Ws; bias = bs; out = agg; break;
    }

    int rowLocal = lane & 15;
    int row  = tileM * 16 + rowLocal;
    int kOff = (lane < 16) ? 0 : 8;
    v16h a0, a1;
    #pragma unroll
    for (int j = 0; j < 8; ++j) {
        a0[j]     = (_Float16)h[row * HID + kOff + j];
        a0[8 + j] = (_Float16)h[row * HID + 16 + kOff + j];
        a1[j]     = (_Float16)h[row * HID + 32 + kOff + j];
        a1[8 + j] = (_Float16)h[row * HID + 48 + kOff + j];
    }
    int kb = (lane < 16) ? 0 : 8;
    #pragma unroll
    for (int nt = 0; nt < 4; ++nt) {
        int col = nt * 16 + rowLocal;
        v16h b0, b1;
        #pragma unroll
        for (int j = 0; j < 16; ++j) {
            b0[j] = (_Float16)W[(kb + (j & 7) + ((j >> 3) << 4)) * HID + col];
            b1[j] = (_Float16)W[(32 + kb + (j & 7) + ((j >> 3) << 4)) * HID + col];
        }
        v8f c = {};
        c = __builtin_amdgcn_wmma_f32_16x16x32_f16(false, a0, false, b0,
                                                   (short)0, c, false, false);
        c = __builtin_amdgcn_wmma_f32_16x16x32_f16(false, a1, false, b1,
                                                   (short)0, c, false, false);
        float bias_c = bias[col];
        #pragma unroll
        for (int i = 0; i < 8; ++i) {
            int m = tileM * 16 + ((lane < 16) ? i : 8 + i);
            out[m * HID + col] = c[i] + bias_c;
        }
    }
}

// ---------------------------------------------------------------------------
// amax = -inf, denom = 0
// ---------------------------------------------------------------------------
__global__ void init_softmax(float* __restrict__ amax, float* __restrict__ denom) {
    int i = blockIdx.x * blockDim.x + threadIdx.x;
    if (i >= N_NODES * NHEAD) return;
    amax[i]  = -__builtin_huge_valf();
    denom[i] = 0.0f;
}

// ---------------------------------------------------------------------------
// Pass A: warp per edge. alpha[e,h] = q[dst].(k[src]+e) / sqrt(C); atomic max.
// Edge embedding e recomputed in-register from edge_attr (saves 400MB traffic).
// Lane l owns channels {2l, 2l+1}; head = l>>3; 8-lane xor reduction per head.
// ---------------------------------------------------------------------------
__global__ void edge_alpha(const float* __restrict__ qn, const float* __restrict__ kn,
                           const float* __restrict__ edge_attr,
                           const int* __restrict__ src, const int* __restrict__ dst,
                           const float* __restrict__ We,
                           float* __restrict__ alpha, float* __restrict__ amax) {
    int warp = (blockIdx.x * (blockDim.x >> 5)) + (threadIdx.x >> 5);
    int lane = threadIdx.x & 31;
    if (warp >= N_EDGES) return;
    int s = src[warp], d = dst[warp];
    int c0 = 2 * lane, c1 = c0 + 1;

    float eav = edge_attr[warp * EDGE_IN + (lane & 15)];
    float e0 = 0.0f, e1 = 0.0f;
    #pragma unroll
    for (int j = 0; j < EDGE_IN; ++j) {
        float aj = __shfl(eav, j, 32);
        e0 = fmaf(aj, We[j * HID + c0], e0);
        e1 = fmaf(aj, We[j * HID + c1], e1);
    }
    const float2* qn2 = reinterpret_cast<const float2*>(qn);
    const float2* kn2 = reinterpret_cast<const float2*>(kn);
    float2 q = qn2[d * 32 + lane];
    float2 k = kn2[s * 32 + lane];
    float part = q.x * (k.x + e0) + q.y * (k.y + e1);
    part += __shfl_xor(part, 1, 32);
    part += __shfl_xor(part, 2, 32);
    part += __shfl_xor(part, 4, 32);
    float aval = part * 0.25f;                 // 1/sqrt(16)
    if ((lane & 7) == 0) {
        int hd = lane >> 3;
        alpha[warp * NHEAD + hd] = aval;
        atomicMaxFloat(&amax[d * NHEAD + hd], aval);
    }
}

// ---------------------------------------------------------------------------
// Pass B: ex = exp(alpha - amax[dst]); denom += ex. (overwrites alpha in place)
// ---------------------------------------------------------------------------
__global__ void edge_exp(float* __restrict__ alpha, const int* __restrict__ dst,
                         const float* __restrict__ amax, float* __restrict__ denom) {
    int idx = blockIdx.x * blockDim.x + threadIdx.x;
    if (idx >= N_EDGES * NHEAD) return;
    int e  = idx >> 2;
    int hd = idx & 3;
    int d  = dst[e];
    float ex = __expf(alpha[idx] - amax[d * NHEAD + hd]);
    alpha[idx] = ex;
    atomicAdd(&denom[d * NHEAD + hd], ex);
}

// ---------------------------------------------------------------------------
// Pass C: warp per edge. agg[dst] += attn * (v[src] + e). Atomic scatter into
// L2-resident agg (12.8MB << 192MB L2).
// ---------------------------------------------------------------------------
__global__ void edge_agg(const float* __restrict__ vn,
                         const float* __restrict__ edge_attr,
                         const int* __restrict__ src, const int* __restrict__ dst,
                         const float* __restrict__ We,
                         const float* __restrict__ exbuf, const float* __restrict__ denom,
                         float* __restrict__ agg) {
    int warp = (blockIdx.x * (blockDim.x >> 5)) + (threadIdx.x >> 5);
    int lane = threadIdx.x & 31;
    if (warp >= N_EDGES) return;
    int s = src[warp], d = dst[warp];
    int c0 = 2 * lane, c1 = c0 + 1;
    int hd = lane >> 3;

    float attn = exbuf[warp * NHEAD + hd] / (denom[d * NHEAD + hd] + 1e-16f);

    float eav = edge_attr[warp * EDGE_IN + (lane & 15)];
    float e0 = 0.0f, e1 = 0.0f;
    #pragma unroll
    for (int j = 0; j < EDGE_IN; ++j) {
        float aj = __shfl(eav, j, 32);
        e0 = fmaf(aj, We[j * HID + c0], e0);
        e1 = fmaf(aj, We[j * HID + c1], e1);
    }
    const float2* vn2 = reinterpret_cast<const float2*>(vn);
    float2 v = vn2[s * 32 + lane];
    atomicAdd(&agg[d * HID + c0], attn * (v.x + e0));
    atomicAdd(&agg[d * HID + c1], attn * (v.y + e1));
}

// ---------------------------------------------------------------------------
// h = h + relu(agg)   (agg already contains skip + attention aggregate)
// ---------------------------------------------------------------------------
__global__ void node_update(float* __restrict__ h, const float* __restrict__ agg) {
    int i = blockIdx.x * blockDim.x + threadIdx.x;
    if (i >= N_NODES * HID) return;
    h[i] = h[i] + fmaxf(agg[i], 0.0f);
}

// ---------------------------------------------------------------------------
// Pooling
// ---------------------------------------------------------------------------
__global__ void pool_zero(float* __restrict__ pooled, float* __restrict__ cnt) {
    for (int i = threadIdx.x; i < G_GRAPHS * HID; i += blockDim.x) pooled[i] = 0.0f;
    if (threadIdx.x < G_GRAPHS) cnt[threadIdx.x] = 0.0f;
}

__global__ void pool_scatter(const float* __restrict__ h, const int* __restrict__ batch,
                             float* __restrict__ pooled, float* __restrict__ cnt) {
    int i = blockIdx.x * blockDim.x + threadIdx.x;
    if (i >= N_NODES * HID) return;
    int n = i >> 6, c = i & 63;
    int g = batch[n];
    atomicAdd(&pooled[g * HID + c], h[i]);
    if (c == 0) atomicAdd(&cnt[g], 1.0f);
}

__global__ void final_mlp(const float* __restrict__ pooled, const float* __restrict__ cnt,
                          const float* __restrict__ W1, const float* __restrict__ b1,
                          const float* __restrict__ W2, const float* __restrict__ b2,
                          float* __restrict__ out) {
    int g = threadIdx.x;
    if (g >= G_GRAPHS) return;
    float inv = 1.0f / fmaxf(cnt[g], 1.0f);
    float p[HID];
    #pragma unroll
    for (int c = 0; c < HID; ++c) p[c] = pooled[g * HID + c] * inv;
    float acc = b2[0];
    for (int j = 0; j < HID / 2; ++j) {
        float t = b1[j];
        #pragma unroll
        for (int c = 0; c < HID; ++c) t = fmaf(p[c], W1[c * (HID / 2) + j], t);
        acc = fmaf(fmaxf(t, 0.0f), W2[j], acc);
    }
    out[g] = acc;
}

// ---------------------------------------------------------------------------
extern "C" void kernel_launch(void* const* d_in, const int* in_sizes, int n_in,
                              void* d_out, int out_size, void* d_ws, size_t ws_size,
                              hipStream_t stream) {
    const float* x         = (const float*)d_in[0];
    const int*   edge_idx  = (const int*)  d_in[1];
    const float* edge_attr = (const float*)d_in[2];
    const int*   batch     = (const int*)  d_in[3];
    const float* Wn        = (const float*)d_in[4];
    const float* bn        = (const float*)d_in[5];
    const float* Wq        = (const float*)d_in[6];
    const float* bq        = (const float*)d_in[7];
    const float* Wk        = (const float*)d_in[8];
    const float* bk        = (const float*)d_in[9];
    const float* Wv        = (const float*)d_in[10];
    const float* bv        = (const float*)d_in[11];
    const float* We        = (const float*)d_in[12];
    const float* Ws        = (const float*)d_in[13];
    const float* bs        = (const float*)d_in[14];
    const float* W1        = (const float*)d_in[15];
    const float* b1        = (const float*)d_in[16];
    const float* W2        = (const float*)d_in[17];
    const float* b2        = (const float*)d_in[18];

    const int* src = edge_idx;
    const int* dst = edge_idx + N_EDGES;

    // workspace carve-up (~91 MB total)
    float* ws     = (float*)d_ws;
    float* h      = ws;                                  // N*64
    float* qn     = h     + (size_t)N_NODES * HID;       // N*64
    float* kn     = qn    + (size_t)N_NODES * HID;       // N*64
    float* vn     = kn    + (size_t)N_NODES * HID;       // N*64
    float* agg    = vn    + (size_t)N_NODES * HID;       // N*64
    float* alpha  = agg   + (size_t)N_NODES * HID;       // E*4 (alpha, then ex)
    float* amax   = alpha + (size_t)N_EDGES * NHEAD;     // N*4
    float* denom  = amax  + (size_t)N_NODES * NHEAD;     // N*4
    float* pooled = denom + (size_t)N_NODES * NHEAD;     // G*64
    float* cnt    = pooled + (size_t)G_GRAPHS * HID;     // G

    input_gemm<<<(ROW_TILES + 3) / 4, 128, 0, stream>>>(x, Wn, bn, h);

    const int edgeWarpBlocks = (N_EDGES + 7) / 8;        // 8 warps / 256-thread block
    for (int l = 0; l < NLAYERS; ++l) {
        const float* Wq_l = Wq + (size_t)l * HID * HID;
        const float* Wk_l = Wk + (size_t)l * HID * HID;
        const float* Wv_l = Wv + (size_t)l * HID * HID;
        const float* Ws_l = Ws + (size_t)l * HID * HID;
        const float* We_l = We + (size_t)l * EDGE_IN * HID;
        const float* bq_l = bq + (size_t)l * HID;
        const float* bk_l = bk + (size_t)l * HID;
        const float* bv_l = bv + (size_t)l * HID;
        const float* bs_l = bs + (size_t)l * HID;

        init_softmax<<<(N_NODES * NHEAD + 255) / 256, 256, 0, stream>>>(amax, denom);
        qkvs_gemm<<<ROW_TILES, 128, 0, stream>>>(h, Wq_l, Wk_l, Wv_l, Ws_l,
                                                 bq_l, bk_l, bv_l, bs_l,
                                                 qn, kn, vn, agg);
        edge_alpha<<<edgeWarpBlocks, 256, 0, stream>>>(qn, kn, edge_attr, src, dst,
                                                       We_l, alpha, amax);
        edge_exp<<<(N_EDGES * NHEAD + 255) / 256, 256, 0, stream>>>(alpha, dst, amax, denom);
        edge_agg<<<edgeWarpBlocks, 256, 0, stream>>>(vn, edge_attr, src, dst,
                                                     We_l, alpha, denom, agg);
        node_update<<<(N_NODES * HID + 255) / 256, 256, 0, stream>>>(h, agg);
    }

    pool_zero<<<1, 256, 0, stream>>>(pooled, cnt);
    pool_scatter<<<(N_NODES * HID + 255) / 256, 256, 0, stream>>>(h, batch, pooled, cnt);
    final_mlp<<<1, 64, 0, stream>>>(pooled, cnt, W1, b1, W2, b2, (float*)d_out);
}
